// NodeEdgeBlock_57019985822460
// MI455X (gfx1250) — compile-verified, matrix operand order checked by reading, not run on hardware
//
#include <hip/hip_runtime.h>
#include <math.h>

// ---------------- types ----------------
typedef __attribute__((ext_vector_type(4)))  float  v4f;
typedef __attribute__((ext_vector_type(8)))  float  v8f;
typedef __attribute__((ext_vector_type(8)))  __bf16 bf8;
typedef __attribute__((ext_vector_type(16))) __bf16 bf16v;

struct KP { const float* in[50]; float* out; char* ws; };

// ---------------- workspace layout (bytes) ----------------
// bf16 weights
constexpr size_t W_INE_OFF  = 0;                       // 256x256 bf16
constexpr size_t W_ADDE_OFF = W_INE_OFF  + 131072;
constexpr size_t W_MULE_OFF = W_ADDE_OFF + 131072;
constexpr size_t W_EOUT_OFF = W_MULE_OFF + 131072;
constexpr size_t W_XE1_OFF  = W_EOUT_OFF + 131072;
constexpr size_t W_XE2_OFF  = W_XE1_OFF  + 131072;
constexpr size_t W_Q_OFF    = W_XE2_OFF  + 131072;
constexpr size_t W_K_OFF    = W_Q_OFF    + 131072;
constexpr size_t W_V_OFF    = W_K_OFF    + 131072;
constexpr size_t W_XOUT_OFF = W_V_OFF    + 131072;
constexpr size_t W_A_OFF    = W_XOUT_OFF + 131072;     // 16x256 (8 rows padded)
constexpr size_t W_EATT_OFF = W_A_OFF    + 8192;
constexpr size_t W_PATT_OFF = W_EATT_OFF + 8192;
constexpr size_t W_OUT_OFF  = W_PATT_OFF + 8192;       // 256x2048
constexpr size_t W_EXM_OFF  = W_OUT_OFF  + 1048576;    // 256x1024
constexpr size_t W_PXM_OFF  = W_EXM_OFF  + 524288;
// f32 node tensors (512x256)
constexpr size_t XM1F_OFF   = W_PXM_OFF  + 524288;
constexpr size_t XM2F_OFF   = XM1F_OFF   + 524288;
constexpr size_t QF_OFF     = XM2F_OFF   + 524288;
constexpr size_t KF_OFF     = QF_OFF     + 524288;
constexpr size_t VT_OFF     = KF_OFF     + 524288;     // bf16 V^T: [b][c][j]
constexpr size_t POSM_OFF   = VT_OFF     + 262144;     // 512x3 f32
constexpr size_t NPOS_OFF   = POSM_OFF   + 6144;
constexpr size_t NORMP_OFF  = NPOS_OFF   + 6144;       // 512 f32
constexpr size_t DENOM_OFF  = NORMP_OFF  + 2048;       // 2 f32 (padded)
constexpr size_t YEA_OFF    = DENOM_OFF  + 256;        // 2x256 f32
constexpr size_t YEM_OFF    = YEA_OFF    + 2048;
constexpr size_t YXM_OFF    = YEM_OFF    + 2048;
constexpr size_t YXA_OFF    = YXM_OFF    + 2048;
constexpr size_t LOGITS_OFF = YXA_OFF    + 2048;       // 2x256x256x8 f32
constexpr size_t STATSE_OFF = LOGITS_OFF + 4194304;    // bf16 512x1024
constexpr size_t STATSD_OFF = STATSE_OFF + 1048576;
constexpr size_t WVFLAT_OFF = STATSD_OFF + 1048576;    // bf16 512x2048
constexpr size_t EOUT_BASE  = 131072;                  // floats, Xout first in d_out

// ---------------- WMMA helpers ----------------
// 16-bit A/B fragment layout (CDNA5 ISA): lane L holds row (L&15);
// element e maps to k = (e/8)*16 + (L>>4)*8 + (e&7)  => two contiguous 16B chunks.
static __device__ __forceinline__ bf16v frag_ld(const __bf16* base, int ld, int r0, int k0) {
  int lane = (int)(threadIdx.x & 31);
  const __bf16* p = base + (size_t)(r0 + (lane & 15)) * (size_t)ld
                         + (size_t)(k0 + ((lane >> 4) << 3));
  bf8 lo = *(const bf8*)(p);
  bf8 hi = *(const bf8*)(p + 16);
  return __builtin_shufflevector(lo, hi, 0,1,2,3,4,5,6,7,8,9,10,11,12,13,14,15);
}

static __device__ __forceinline__ v8f wmma_bf(bf16v a, bf16v b, v8f c) {
  return __builtin_amdgcn_wmma_f32_16x16x32_bf16(false, a, false, b, (short)0, c, false, false);
}

// ======================================================================
// K1: weight conversion, per-node pos terms, y-conditioning linears
// ======================================================================
__global__ __launch_bounds__(256) void k1_prep(KP P) {
  char* ws = P.ws;
  int gt = blockIdx.x * 256 + threadIdx.x;
  const int gsz = 64 * 256;
  // plain f32 -> bf16 weight conversions
  {
    const int    srcs[13] = {5,17,19,48,7,9,23,21,25,46,28,34,36};
    const size_t offs[13] = {W_INE_OFF,W_ADDE_OFF,W_MULE_OFF,W_EOUT_OFF,W_XE1_OFF,W_XE2_OFF,
                             W_Q_OFF,W_K_OFF,W_V_OFF,W_XOUT_OFF,W_OUT_OFF,W_EXM_OFF,W_PXM_OFF};
    const int    ns[13]   = {65536,65536,65536,65536,65536,65536,65536,65536,65536,65536,
                             524288,262144,262144};
    for (int jj = 0; jj < 13; jj++) {
      const float* s = P.in[srcs[jj]];
      __bf16* d = (__bf16*)(ws + offs[jj]);
      for (int idx = gt; idx < ns[jj]; idx += gsz) d[idx] = (__bf16)s[idx];
    }
  }
  // 8-row matrices padded to 16 rows (rows 8..15 = 0)
  {
    const int    srcs[3] = {27,30,32};
    const size_t offs[3] = {W_A_OFF, W_EATT_OFF, W_PATT_OFF};
    for (int jj = 0; jj < 3; jj++) {
      const float* s = P.in[srcs[jj]];
      __bf16* d = (__bf16*)(ws + offs[jj]);
      for (int idx = gt; idx < 4096; idx += gsz)
        d[idx] = ((idx >> 8) < 8) ? (__bf16)s[idx] : (__bf16)0.0f;
    }
  }
  // per-node: masked pos, norm, normalized pos
  {
    const float* pos = P.in[3]; const float* msk = P.in[4];
    float* posm  = (float*)(ws + POSM_OFF);
    float* npos  = (float*)(ws + NPOS_OFF);
    float* normp = (float*)(ws + NORMP_OFF);
    for (int idx = gt; idx < 512; idx += gsz) {
      float mi = msk[idx];
      float px = pos[idx*3+0]*mi, py = pos[idx*3+1]*mi, pz = pos[idx*3+2]*mi;
      float s2 = px*px + py*py + pz*pz;
      float nr = (s2 > 0.f) ? sqrtf(s2) : 0.f;
      float iv = 1.f / (nr + 1e-7f);
      posm[idx*3+0]=px; posm[idx*3+1]=py; posm[idx*3+2]=pz;
      npos[idx*3+0]=px*iv; npos[idx*3+1]=py*iv; npos[idx*3+2]=pz*iv;
      normp[idx] = nr;
    }
  }
  if (gt < 2) { // mask denominators
    float s = 0.f;
    for (int j = 0; j < 256; j++) s += P.in[4][gt*256 + j];
    ((float*)(ws + DENOM_OFF))[gt] = s;
  }
  // y linears: y_e_mul, y_e_add, y_x_mul, y_x_add  (2x256 each, K=128)
  {
    const int    wsel[4] = {38,40,42,44};
    const size_t dof[4]  = {YEM_OFF, YEA_OFF, YXM_OFF, YXA_OFF};
    const float* y = P.in[2];
    for (int idx = gt; idx < 2048; idx += gsz) {
      int sel = idx >> 9, rem = idx & 511, b = rem >> 8, c = rem & 255;
      const float* w = P.in[wsel[sel]];
      float acc = P.in[wsel[sel] + 1][c];
      for (int d = 0; d < 128; d++) acc += y[b*128 + d] * w[c*128 + d];
      ((float*)(ws + dof[sel]))[rem] = acc;
    }
  }
}

// ======================================================================
// K2: node GEMMs  (xm1, xm2, Q, K, V^T) : 512x256 @ 256x256
// ======================================================================
__global__ __launch_bounds__(256) void k2_node(KP P) {
  __shared__ __bf16 Xbf[64 * 256];
  char* ws = P.ws;
  int tid = threadIdx.x;
  int r0 = blockIdx.x * 64;
  for (int it = tid; it < 4096; it += 256) {     // stage X tile -> bf16 LDS
    int rr = it >> 6, cc = (it & 63) * 4;
    v4f x = *(const v4f*)(P.in[0] + (size_t)(r0 + rr) * 256 + cc);
    Xbf[rr*256+cc+0] = (__bf16)x[0]; Xbf[rr*256+cc+1] = (__bf16)x[1];
    Xbf[rr*256+cc+2] = (__bf16)x[2]; Xbf[rr*256+cc+3] = (__bf16)x[3];
  }
  __syncthreads();
  int lane = tid & 31, wvi = tid >> 5;
  int mt = wvi & 3, nh = wvi >> 2;
  int rbase = (lane >> 4) << 3;
  const size_t woff[5] = {W_XE1_OFF, W_XE2_OFF, W_Q_OFF, W_K_OFF, W_V_OFF};
  const int    bidx[5] = {8, 10, 24, 22, 26};
  const size_t doff[4] = {XM1F_OFF, XM2F_OFF, QF_OFF, KF_OFF};
  for (int wi = 0; wi < 5; wi++) {
    const __bf16* wb = (const __bf16*)(ws + woff[wi]);
    const float* bias = P.in[bidx[wi]];
    for (int nt8 = 0; nt8 < 8; nt8++) {
      int nt = nh * 8 + nt8;
      v8f acc = {};
      for (int k0 = 0; k0 < 256; k0 += 32) {
        bf16v a = frag_ld(Xbf, 256, mt * 16, k0);
        bf16v b = frag_ld(wb,  256, nt * 16, k0);
        acc = wmma_bf(a, b, acc);
      }
      int c = nt * 16 + (lane & 15);
      float bc = bias[c];
      for (int r = 0; r < 8; r++) {
        int gr = r0 + mt * 16 + rbase + r;
        float mi = P.in[4][gr];
        float val = (acc[r] + bc) * mi;
        if (wi < 4) ((float*)(ws + doff[wi]))[(size_t)gr * 256 + c] = val;
        else {
          int bb = gr >> 8, ii = gr & 255;   // V^T: [b][c][j]
          ((__bf16*)(ws + VT_OFF))[((size_t)bb * 256 + c) * 256 + ii] = (__bf16)val;
        }
      }
    }
  }
}

// ======================================================================
// K3: fused edge kernel — one block per (b,i)
// ======================================================================
constexpr int K3_SMEM = 111360;
__global__ __launch_bounds__(256) void k3_edge(KP P) {
  extern __shared__ char smem[];
  __bf16* EBF  = (__bf16*)(smem + 0);        // 32x256 bf16
  __bf16* DBF  = (__bf16*)(smem + 16384);    // dist1
  __bf16* PBF  = (__bf16*)(smem + 32768);    // Q*K/16
  __bf16* ENBF = (__bf16*)(smem + 49152);    // Enew
  float*  XM1T = (float*)(smem + 65536);     // 32x256 f32
  float*  QI   = (float*)(smem + 98304);
  float*  XM2I = (float*)(smem + 99328);
  float*  A0   = (float*)(smem + 100352);
  float*  A1   = (float*)(smem + 101376);
  float*  A2   = (float*)(smem + 102400);
  float*  A3   = (float*)(smem + 103424);
  float*  YEAI = (float*)(smem + 104448);
  float*  YEMI = (float*)(smem + 105472);
  float*  BINE = (float*)(smem + 106496);
  float*  BADD = (float*)(smem + 107520);
  float*  BMUL = (float*)(smem + 108544);
  float*  BEOUT= (float*)(smem + 109568);
  float*  MJ   = (float*)(smem + 110592);
  float*  DISTV= (float*)(smem + 110720);
  float*  COSV = (float*)(smem + 110848);
  float*  NPJ  = (float*)(smem + 110976);
  float*  POSI = (float*)(smem + 111104);

  char* ws = P.ws;
  int tid = threadIdx.x;
  int b = blockIdx.x >> 8, node = blockIdx.x;
  const __bf16* wIne  = (const __bf16*)(ws + W_INE_OFF);
  const __bf16* wAdd  = (const __bf16*)(ws + W_ADDE_OFF);
  const __bf16* wMul  = (const __bf16*)(ws + W_MULE_OFF);
  const __bf16* wEout = (const __bf16*)(ws + W_EOUT_OFF);
  const __bf16* wA    = (const __bf16*)(ws + W_A_OFF);
  const __bf16* wEatt = (const __bf16*)(ws + W_EATT_OFF);
  const __bf16* wPatt = (const __bf16*)(ws + W_PATT_OFF);

  { // per-block init: i-side vectors + channel constants
    int c = tid;
    QI[c]   = ((const float*)(ws + QF_OFF))[(size_t)node * 256 + c];
    XM2I[c] = ((const float*)(ws + XM2F_OFF))[(size_t)node * 256 + c];
    float npI = ((const float*)(ws + NORMP_OFF))[node];
    A0[c] = P.in[11][c*2+0];                 // lin_dist1 w[:,0]
    A1[c] = P.in[11][c*2+1];                 // lin_dist1 w[:,1]
    A2[c] = P.in[15][c];                     // lin_norm_pos2 w
    A3[c] = P.in[12][c] + P.in[13][c]*npI + P.in[14][c] + P.in[16][c];
    YEAI[c] = ((const float*)(ws + YEA_OFF))[b*256 + c];
    YEMI[c] = ((const float*)(ws + YEM_OFF))[b*256 + c];
    BINE[c] = P.in[6][c];  BADD[c] = P.in[18][c];
    BMUL[c] = P.in[20][c]; BEOUT[c] = P.in[49][c];
    if (tid == 0) {
      const float* pm = (const float*)(ws + POSM_OFF);
      const float* np = (const float*)(ws + NPOS_OFF);
      POSI[0]=pm[node*3+0]; POSI[1]=pm[node*3+1]; POSI[2]=pm[node*3+2];
      POSI[3]=np[node*3+0]; POSI[4]=np[node*3+1]; POSI[5]=np[node*3+2];
      POSI[6]=P.in[4][node];
    }
  }
  __syncthreads();

  float sE=0,qE=0,mnE=3.0e38f,mxE=-3.0e38f;
  float sD=0,qD=0,mnD=3.0e38f,mxD=-3.0e38f;
  int lane = tid & 31, wvi = tid >> 5;
  int rbase = (lane >> 4) << 3;
  int mt = wvi & 1, ng = wvi >> 1;
  int row = tid >> 3, seg = (tid & 7) * 32;

  for (int jt = 0; jt < 8; jt++) {
    int j0 = jt * 32;
    { // stage E (f32->bf16), xm1 (f32), prod=Q*K/16 (bf16)
      const float* esrc = P.in[1] + (((size_t)node * 256 + j0 + row) * 256 + seg);
      __bf16* edst = EBF + row * 256 + seg;
      const float* xs = ((const float*)(ws + XM1F_OFF)) + ((size_t)b*256 + j0 + row)*256 + seg;
      float* xd = XM1T + row * 256 + seg;
      const float* ksrc = ((const float*)(ws + KF_OFF)) + ((size_t)b*256 + j0 + row)*256 + seg;
      __bf16* pd = PBF + row * 256 + seg;
      if (jt < 7) __builtin_prefetch(esrc + 32 * 256, 0, 0);   // next j-tile -> global_prefetch_b8
      for (int q8 = 0; q8 < 4; q8++) {
        v4f x0 = *(const v4f*)(esrc + q8*8);
        v4f x1 = *(const v4f*)(esrc + q8*8 + 4);
        bf8 o;
        o[0]=(__bf16)x0[0]; o[1]=(__bf16)x0[1]; o[2]=(__bf16)x0[2]; o[3]=(__bf16)x0[3];
        o[4]=(__bf16)x1[0]; o[5]=(__bf16)x1[1]; o[6]=(__bf16)x1[2]; o[7]=(__bf16)x1[3];
        *(bf8*)(edst + q8*8) = o;
        *(v4f*)(xd + q8*8)     = *(const v4f*)(xs + q8*8);
        *(v4f*)(xd + q8*8 + 4) = *(const v4f*)(xs + q8*8 + 4);
        v4f k0v = *(const v4f*)(ksrc + q8*8);
        v4f k1v = *(const v4f*)(ksrc + q8*8 + 4);
        int cb = seg + q8*8;
        bf8 pr;
        pr[0]=(__bf16)(QI[cb+0]*k0v[0]*0.0625f); pr[1]=(__bf16)(QI[cb+1]*k0v[1]*0.0625f);
        pr[2]=(__bf16)(QI[cb+2]*k0v[2]*0.0625f); pr[3]=(__bf16)(QI[cb+3]*k0v[3]*0.0625f);
        pr[4]=(__bf16)(QI[cb+4]*k1v[0]*0.0625f); pr[5]=(__bf16)(QI[cb+5]*k1v[1]*0.0625f);
        pr[6]=(__bf16)(QI[cb+6]*k1v[2]*0.0625f); pr[7]=(__bf16)(QI[cb+7]*k1v[3]*0.0625f);
        *(bf8*)(pd + q8*8) = pr;
      }
      if (tid < 32) { // per-j scalars
        int j = j0 + tid;
        MJ[tid] = P.in[4][b*256 + j];
        const float* pj = ((const float*)(ws + POSM_OFF)) + (size_t)(b*256 + j) * 3;
        float dx = POSI[0]-pj[0], dy = POSI[1]-pj[1], dz = POSI[2]-pj[2];
        float d2 = dx*dx + dy*dy + dz*dz;
        DISTV[tid] = (d2 > 0.f) ? sqrtf(d2) : 0.f;
        const float* nj = ((const float*)(ws + NPOS_OFF)) + (size_t)(b*256 + j) * 3;
        COSV[tid] = POSI[3]*nj[0] + POSI[4]*nj[1] + POSI[5]*nj[2];
        NPJ[tid]  = ((const float*)(ws + NORMP_OFF))[b*256 + j];
      }
    }
    __syncthreads();
    { // dist1 tile (analytic, rank-1 norm terms), relu * mask
      float mm = POSI[6] * MJ[row];
      float dv = DISTV[row], cv = COSV[row], nv = NPJ[row];
      for (int q = 0; q < 32; q++) {
        int c = seg + q;
        float pre = A0[c]*dv + A1[c]*cv + A2[c]*nv + A3[c];
        DBF[row*256 + c] = (__bf16)(fmaxf(pre, 0.f) * mm);
      }
    }
    __syncthreads();
    float MI = POSI[6];
    // round 1: in_E / dist_add_e / dist_mul_e  ->  Enew (bf16 LDS)
    for (int tt = 0; tt < 4; tt++) {
      int nt = ng * 4 + tt;
      v8f accY = {}, accA = {}, accM = {};
      for (int k0 = 0; k0 < 256; k0 += 32) {
        bf16v aE = frag_ld(EBF, 256, mt*16, k0);
        bf16v aD = frag_ld(DBF, 256, mt*16, k0);
        accY = wmma_bf(aE, frag_ld(wIne, 256, nt*16, k0), accY);
        accA = wmma_bf(aD, frag_ld(wAdd, 256, nt*16, k0), accA);
        accM = wmma_bf(aD, frag_ld(wMul, 256, nt*16, k0), accM);
      }
      int c = nt*16 + (lane & 15);
      float bine=BINE[c], badd=BADD[c], bmul=BMUL[c];
      float xm2c=XM2I[c], yeac=YEAI[c], yemc=YEMI[c];
      for (int r = 0; r < 8; r++) {
        int m = mt*16 + rbase + r;
        float mm = MI * MJ[m];
        float y1 = (accY[r] + bine) * XM1T[m*256 + c] * xm2c * mm;
        float y2 = (y1 + (accA[r] + badd) + y1 * (accM[r] + bmul)) * mm;
        float en = (y2 + yeac + y2 * yemc) * mm;
        ENBF[m*256 + c] = (__bf16)en;
      }
    }
    __syncthreads();
    // round 2: Eout = e_out(Enew) -> global
    for (int tt = 0; tt < 4; tt++) {
      int nt = ng * 4 + tt;
      v8f accO = {};
      for (int k0 = 0; k0 < 256; k0 += 32)
        accO = wmma_bf(frag_ld(ENBF, 256, mt*16, k0), frag_ld(wEout, 256, nt*16, k0), accO);
      int c = nt*16 + (lane & 15);
      float bo = BEOUT[c];
      for (int r = 0; r < 8; r++) {
        int m = mt*16 + rbase + r;
        float mm = MI * MJ[m];
        P.out[EOUT_BASE + ((size_t)node*256 + (j0+m))*256 + c] = (accO[r] + bo) * mm;
      }
    }
    // attention logits: a = (prod@Wa)*(1+Enew@We+be)*(1+dist1@Wp+bp)*mask
    if (wvi < 2) {
      v8f accQ = {}, accE = {}, accP = {};
      for (int k0 = 0; k0 < 256; k0 += 32) {
        accQ = wmma_bf(frag_ld(PBF,  256, wvi*16, k0), frag_ld(wA,    256, 0, k0), accQ);
        accE = wmma_bf(frag_ld(ENBF, 256, wvi*16, k0), frag_ld(wEatt, 256, 0, k0), accE);
        accP = wmma_bf(frag_ld(DBF,  256, wvi*16, k0), frag_ld(wPatt, 256, 0, k0), accP);
      }
      int h = lane & 15;
      if (h < 8) {
        float bea = P.in[31][h], bpa = P.in[33][h];
        for (int r = 0; r < 8; r++) {
          int m = wvi*16 + rbase + r;
          float mm = MI * MJ[m];
          float a0 = accQ[r] * mm;
          float af = a0 * (1.f + accE[r] + bea) * (1.f + accP[r] + bpa) * mm;
          ((float*)(ws + LOGITS_OFF))[((size_t)node*256 + (j0+m))*8 + h] = af;
        }
      }
    }
    { // row stats for _eto_x (thread == channel)
      for (int m = 0; m < 32; m++) {
        float mj = MJ[m], off = 100000.f * (1.f - mj);
        float e = (float)ENBF[m*256 + tid];
        sE += e; qE += e*e; mnE = fminf(mnE, e + off); mxE = fmaxf(mxE, e - off);
        float d = (float)DBF[m*256 + tid];
        sD += d; qD += d*d; mnD = fminf(mnD, d + off); mxD = fmaxf(mxD, d - off);
      }
    }
    __syncthreads();
  }
  { // stats -> (m, min, max, std) bf16, concat layout for e_x_mul / pos_x_mul
    float den = ((const float*)(ws + DENOM_OFF))[b];
    float mE = sE/den, stEv = qE/den - mE*mE;
    float mD = sD/den, stDv = qD/den - mD*mD;
    __bf16* se = (__bf16*)(ws + STATSE_OFF) + (size_t)node * 1024;
    __bf16* sd = (__bf16*)(ws + STATSD_OFF) + (size_t)node * 1024;
    se[tid]=(__bf16)mE; se[256+tid]=(__bf16)mnE; se[512+tid]=(__bf16)mxE; se[768+tid]=(__bf16)stEv;
    sd[tid]=(__bf16)mD; sd[256+tid]=(__bf16)mnD; sd[512+tid]=(__bf16)mxD; sd[768+tid]=(__bf16)stDv;
  }
}

// ======================================================================
// K4: masked softmax over j + alpha @ V  (one block per (b,i))
// ======================================================================
__global__ __launch_bounds__(256) void k4_attn(KP P) {
  __shared__ float  LL[256 * 8];
  __shared__ __bf16 ALPHA[16 * 256];
  char* ws = P.ws;
  int tid = threadIdx.x;
  int node = blockIdx.x, b = node >> 8;
  {
    const float* src = ((const float*)(ws + LOGITS_OFF)) + (size_t)node * 2048 + (size_t)tid * 8;
    float mj = P.in[4][b*256 + tid];
    for (int q = 0; q < 8; q++) LL[tid*8 + q] = (mj > 0.f) ? src[q] : -3.0e38f;
  }
  __syncthreads();
  int lane = tid & 31, wvi = tid >> 5;
  { // wave w owns head h=w
    int h = wvi;
    float mx = -3.0e38f;
    for (int k = 0; k < 8; k++) mx = fmaxf(mx, LL[(lane + 32*k)*8 + h]);
    for (int o = 16; o > 0; o >>= 1) mx = fmaxf(mx, __shfl_xor(mx, o, 32));
    float s = 0.f;
    for (int k = 0; k < 8; k++) {
      float a = LL[(lane + 32*k)*8 + h];
      if (a > -1e37f) s += __expf(a - mx);
    }
    for (int o = 16; o > 0; o >>= 1) s += __shfl_xor(s, o, 32);
    float inv = (s > 0.f) ? 1.f/s : 0.f;
    for (int k = 0; k < 8; k++) {
      int j = lane + 32*k;
      float a = LL[j*8 + h];
      ALPHA[h*256 + j] = (__bf16)((a > -1e37f) ? __expf(a - mx) * inv : 0.f);
      ALPHA[(h+8)*256 + j] = (__bf16)0.0f;   // pad rows
    }
  }
  __syncthreads();
  const __bf16* vt = (const __bf16*)(ws + VT_OFF) + (size_t)b * 65536;
  __bf16* wvf = (__bf16*)(ws + WVFLAT_OFF) + (size_t)node * 2048;
  int rbase = (lane >> 4) << 3;
  for (int tt = 0; tt < 2; tt++) {
    int nt = wvi * 2 + tt;
    v8f acc = {};
    for (int k0 = 0; k0 < 256; k0 += 32)
      acc = wmma_bf(frag_ld(ALPHA, 256, 0, k0), frag_ld(vt, 256, nt*16, k0), acc);
    int c = nt*16 + (lane & 15);
    for (int r = 0; r < 8; r++) {
      int h = rbase + r;
      if (h < 8) wvf[(size_t)h*256 + c] = (__bf16)acc[r];
    }
  }
}

// ======================================================================
// K5: out(K=2048) -> e_x_mul/pos_x_mul(K=1024) -> y-cond -> x_out -> Xout
// ======================================================================
__global__ __launch_bounds__(256) void k5_nodeout(KP P) {
  __shared__ __bf16 NEWX[64 * 256];
  char* ws = P.ws;
  int tid = threadIdx.x;
  int r0 = blockIdx.x * 64;
  int lane = tid & 31, wvi = tid >> 5;
  int mt = wvi & 3, nh = wvi >> 2;
  int rbase = (lane >> 4) << 3;
  v8f res[8];
  const __bf16* wvf  = (const __bf16*)(ws + WVFLAT_OFF) + (size_t)r0 * 2048;
  const __bf16* wOut = (const __bf16*)(ws + W_OUT_OFF);
  for (int nt8 = 0; nt8 < 8; nt8++) {               // 'out' linear
    int nt = nh*8 + nt8;
    v8f acc = {};
    for (int k0 = 0; k0 < 2048; k0 += 32)
      acc = wmma_bf(frag_ld(wvf, 2048, mt*16, k0), frag_ld(wOut, 2048, nt*16, k0), acc);
    int c = nt*16 + (lane & 15);
    float bc = P.in[29][c];
    for (int r = 0; r < 8; r++) {
      int gr = r0 + mt*16 + rbase + r;
      acc[r] = (acc[r] + bc) * P.in[4][gr];
    }
    res[nt8] = acc;
  }
  const __bf16* stE  = (const __bf16*)(ws + STATSE_OFF) + (size_t)r0 * 1024;
  const __bf16* wExm = (const __bf16*)(ws + W_EXM_OFF);
  for (int nt8 = 0; nt8 < 8; nt8++) {               // wV *= (1 + e_x_mul(statsE))
    int nt = nh*8 + nt8;
    v8f acc = {};
    for (int k0 = 0; k0 < 1024; k0 += 32)
      acc = wmma_bf(frag_ld(stE, 1024, mt*16, k0), frag_ld(wExm, 1024, nt*16, k0), acc);
    int c = nt*16 + (lane & 15);
    float bc = P.in[35][c];
    for (int r = 0; r < 8; r++) res[nt8][r] = res[nt8][r] * (1.f + acc[r] + bc);
  }
  const __bf16* stD  = (const __bf16*)(ws + STATSD_OFF) + (size_t)r0 * 1024;
  const __bf16* wPxm = (const __bf16*)(ws + W_PXM_OFF);
  for (int nt8 = 0; nt8 < 8; nt8++) {               // wV *= (1 + pos_x_mul(statsD))
    int nt = nh*8 + nt8;
    v8f acc = {};
    for (int k0 = 0; k0 < 1024; k0 += 32)
      acc = wmma_bf(frag_ld(stD, 1024, mt*16, k0), frag_ld(wPxm, 1024, nt*16, k0), acc);
    int c = nt*16 + (lane & 15);
    float bc = P.in[37][c];
    for (int r = 0; r < 8; r++) res[nt8][r] = res[nt8][r] * (1.f + acc[r] + bc);
  }
  for (int nt8 = 0; nt8 < 8; nt8++) {               // newX = wV*(y_x_mul+1)+y_x_add
    int nt = nh*8 + nt8;
    int c = nt*16 + (lane & 15);
    for (int r = 0; r < 8; r++) {
      int ml = mt*16 + rbase + r;
      int bb = (r0 + ml) >> 8;
      float v = res[nt8][r] * (((const float*)(ws + YXM_OFF))[bb*256 + c] + 1.f)
              + ((const float*)(ws + YXA_OFF))[bb*256 + c];
      NEWX[ml*256 + c] = (__bf16)v;
    }
  }
  __syncthreads();
  const __bf16* wXout = (const __bf16*)(ws + W_XOUT_OFF);
  for (int nt8 = 0; nt8 < 8; nt8++) {               // Xout = x_out(newX)*mask
    int nt = nh*8 + nt8;
    v8f acc = {};
    for (int k0 = 0; k0 < 256; k0 += 32)
      acc = wmma_bf(frag_ld(NEWX, 256, mt*16, k0), frag_ld(wXout, 256, nt*16, k0), acc);
    int c = nt*16 + (lane & 15);
    float bc = P.in[47][c];
    for (int r = 0; r < 8; r++) {
      int gr = r0 + mt*16 + rbase + r;
      P.out[(size_t)gr*256 + c] = (acc[r] + bc) * P.in[4][gr];
    }
  }
}

// ======================================================================
extern "C" void kernel_launch(void* const* d_in, const int* in_sizes, int n_in,
                              void* d_out, int out_size, void* d_ws, size_t ws_size,
                              hipStream_t stream) {
  KP P;
  for (int i = 0; i < 50; i++) P.in[i] = (i < n_in) ? (const float*)d_in[i] : nullptr;
  P.out = (float*)d_out;
  P.ws  = (char*)d_ws;
  k1_prep   <<<64,  256, 0,       stream>>>(P);
  k2_node   <<<8,   256, 0,       stream>>>(P);
  k3_edge   <<<512, 256, K3_SMEM, stream>>>(P);
  k4_attn   <<<512, 256, 0,       stream>>>(P);
  k5_nodeout<<<8,   256, 0,       stream>>>(P);
}
